// GroupedQueryAttention_1554778161505
// MI455X (gfx1250) — matrix operand order repaired and checked by reading, not compiled, not run on hardware
//
#include <hip/hip_runtime.h>
#include <stdint.h>

#define NH   32
#define NKV  8
#define HD   64
#define SEQ  2048
#define EMB  2048
#define BATCH 2

typedef __attribute__((ext_vector_type(16))) __bf16 v16bf;
typedef __attribute__((ext_vector_type(8)))  float  v8f;

union FragBF { uint4 u[2]; v16bf v; };

__device__ __forceinline__ unsigned short f2bf(float f) {
  unsigned u = __float_as_uint(f);
  u += 0x7FFFu + ((u >> 16) & 1u);      // round-to-nearest-even
  return (unsigned short)(u >> 16);
}

// ---------------------------------------------------------------- convert ---
__global__ void cvt_f32_bf16(const float* __restrict__ src,
                             unsigned short* __restrict__ dst, int n) {
  int i = blockIdx.x * blockDim.x + threadIdx.x;
  int stride = gridDim.x * blockDim.x;
  for (; i < n; i += stride) dst[i] = f2bf(src[i]);
}

// ------------------------------------------------------------------- GEMM ---
// C[M,N] = A[M,K](bf16,row-major) * W[N,K]^T (bf16,row-major)  ("NT" gemm)
// mode 0: store f32 plain to outF[M,N]
// mode 1: fused RoPE, store bf16 to outB[b][head][s][d]   (Q / K)
// mode 2: no rope, store bf16 TRANSPOSED to outB[b][head][d][s]  (V)

// compute one k-stage (K=32) of the 32x64 wave tile from LDS buffer BUFOFF
#define GEMM_COMPUTE(BUFOFF)                                                   \
  do {                                                                         \
    FragBF af[2], bfr[4];                                                      \
    _Pragma("unroll")                                                          \
    for (int mt = 0; mt < 2; ++mt) {                                           \
      const int ix = (BUFOFF) + (wm + mt * 16 + nl) * 32 + half * 8;           \
      af[mt].u[0] = *reinterpret_cast<const uint4*>(&As[ix]);                  \
      af[mt].u[1] = *reinterpret_cast<const uint4*>(&As[ix + 16]);             \
    }                                                                          \
    _Pragma("unroll")                                                          \
    for (int nt = 0; nt < 4; ++nt) {                                           \
      const int ix = (BUFOFF) + (wn + nt * 16 + nl) * 32 + half * 8;           \
      bfr[nt].u[0] = *reinterpret_cast<const uint4*>(&Bs[ix]);                 \
      bfr[nt].u[1] = *reinterpret_cast<const uint4*>(&Bs[ix + 16]);            \
    }                                                                          \
    _Pragma("unroll")                                                          \
    for (int mt = 0; mt < 2; ++mt)                                             \
      _Pragma("unroll")                                                        \
      for (int nt = 0; nt < 4; ++nt)                                           \
        acc[mt][nt] = __builtin_amdgcn_wmma_f32_16x16x32_bf16(                 \
            false, af[mt].v, false, bfr[nt].v, (short)0, acc[mt][nt],          \
            false, false);                                                     \
  } while (0)

__global__ __launch_bounds__(256)
void gemm_bf16_nt(const unsigned short* __restrict__ A,
                  const unsigned short* __restrict__ W,
                  float* __restrict__ outF,
                  unsigned short* __restrict__ outB,
                  const float* __restrict__ cosT,
                  const float* __restrict__ sinT,
                  int M, int N, int K, int mode, int nheads)
{
  __shared__ unsigned short As[2 * 128 * 32];
  __shared__ unsigned short Bs[2 * 128 * 32];

  const int tid  = threadIdx.x;
  const int wave = tid >> 5, lane = tid & 31;
  const int half = lane >> 4, nl = lane & 15;
  const int wm = (wave >> 1) * 32;   // wave row offset inside 128 block
  const int wn = (wave & 1) * 64;    // wave col offset inside 128 block
  const int bm = blockIdx.y * 128, bn = blockIdx.x * 128;

  // staging: each thread moves two 8-bf16 chunks for A and two for B per stage
  const int rS = tid >> 2, cS = (tid & 3) * 8;          // row 0..63, col chunk
  const size_t rowStep = (size_t)64 * K;                // +64 rows
  const unsigned short* aG = A + (size_t)(bm + rS) * K + cS;
  const unsigned short* bG = W + (size_t)(bn + rS) * K + cS;
  const int lo = rS * 32 + cS;                          // LDS offset, buffer 0

  // ---- prologue: stage 0 ----
  {
    uint4 a0 = *reinterpret_cast<const uint4*>(aG);
    uint4 a1 = *reinterpret_cast<const uint4*>(aG + rowStep);
    uint4 b0 = *reinterpret_cast<const uint4*>(bG);
    uint4 b1 = *reinterpret_cast<const uint4*>(bG + rowStep);
    *reinterpret_cast<uint4*>(&As[lo])            = a0;
    *reinterpret_cast<uint4*>(&As[lo + 64 * 32])  = a1;
    *reinterpret_cast<uint4*>(&Bs[lo])            = b0;
    *reinterpret_cast<uint4*>(&Bs[lo + 64 * 32])  = b1;
  }
  __syncthreads();

  v8f acc[2][4];
#pragma unroll
  for (int mt = 0; mt < 2; ++mt)
#pragma unroll
    for (int nt = 0; nt < 4; ++nt)
#pragma unroll
      for (int r = 0; r < 8; ++r) acc[mt][nt][r] = 0.f;

  const int nk = K >> 5;
  for (int k = 0; k < nk - 1; ++k) {
    const int cur  = k & 1;
    const int koff = (k + 1) << 5;
    // load next stage (short live range -> stays in VGPRs)
    uint4 a0 = *reinterpret_cast<const uint4*>(aG + koff);
    uint4 a1 = *reinterpret_cast<const uint4*>(aG + koff + rowStep);
    uint4 b0 = *reinterpret_cast<const uint4*>(bG + koff);
    uint4 b1 = *reinterpret_cast<const uint4*>(bG + koff + rowStep);
    if (k + 2 < nk) {                 // warm L2 two stages ahead
      __builtin_prefetch(aG + ((k + 2) << 5), 0, 0);
      __builtin_prefetch(bG + ((k + 2) << 5), 0, 0);
    }

    GEMM_COMPUTE(cur * 4096);

    const int so = (cur ^ 1) * 4096 + lo;
    *reinterpret_cast<uint4*>(&As[so])           = a0;
    *reinterpret_cast<uint4*>(&As[so + 64 * 32]) = a1;
    *reinterpret_cast<uint4*>(&Bs[so])           = b0;
    *reinterpret_cast<uint4*>(&Bs[so + 64 * 32]) = b1;
    __syncthreads();
  }
  GEMM_COMPUTE(((nk - 1) & 1) * 4096);

  // ---- epilogue ----
#pragma unroll
  for (int mt = 0; mt < 2; ++mt) {
#pragma unroll
    for (int nt = 0; nt < 4; ++nt) {
#pragma unroll
      for (int r = 0; r < 8; ++r) {
        float v = acc[mt][nt][r];
        int row = bm + wm + mt * 16 + r + 8 * half;
        int col = bn + wn + nt * 16 + nl;
        if (mode == 0) {
          outF[(size_t)row * N + col] = v;
        } else {
          int h = col >> 6, d = col & 63;
          int bb = row >> 11, s = row & (SEQ - 1);
          if (mode == 1) {
            float c  = cosT[s * HD + d];
            float sn = sinT[s * HD + d];
            float partner = acc[mt][nt ^ 2][r];   // d +/- 32 lives in tile nt^2
            float o = (d < 32) ? (v * c - partner * sn) : (v * c + partner * sn);
            outB[(((size_t)(bb * nheads + h)) * SEQ + s) * HD + d] = f2bf(o);
          } else {
            // V: transposed store [b][kv][d][s] so PV B-fragments are contiguous
            outB[(((size_t)(bb * nheads + h)) * HD + d) * SEQ + s] = f2bf(v);
          }
        }
      }
    }
  }
}

// -------------------------------------------------------- flash attention ---
__device__ __forceinline__ float rmax16(float v) {
#pragma unroll
  for (int m = 1; m < 16; m <<= 1) v = fmaxf(v, __shfl_xor(v, m, 32));
  return v;
}
__device__ __forceinline__ float rsum16(float v) {
#pragma unroll
  for (int m = 1; m < 16; m <<= 1) v += __shfl_xor(v, m, 32);
  return v;
}

__global__ __launch_bounds__(128)
void attn_kernel(const unsigned short* __restrict__ qh,
                 const unsigned short* __restrict__ kh,
                 const unsigned short* __restrict__ vh,
                 unsigned short* __restrict__ ctxb)
{
  __shared__ unsigned short plds[4 * 16 * 64];   // wave-private P tiles

  const int qb = blockIdx.x, h = blockIdx.y, b = blockIdx.z;
  const int kv = h >> 2;                          // NH/NKV = 4
  const int tid = threadIdx.x, wave = tid >> 5, lane = tid & 31;
  const int half = lane >> 4, nl = lane & 15;
  const float scale = 0.125f;                     // 1/sqrt(HD)

  const unsigned short* Q  = qh + (size_t)(b * NH  + h ) * SEQ * HD;
  const unsigned short* Kp = kh + (size_t)(b * NKV + kv) * SEQ * HD;
  const unsigned short* Vt = vh + (size_t)(b * NKV + kv) * HD * SEQ;
  unsigned short* pw = plds + wave * (16 * 64);
  const int q0 = qb * 64 + wave * 16;

  // Q fragments (A layout) loaded straight from global — stays resident
  FragBF aq[2];
#pragma unroll
  for (int kc2 = 0; kc2 < 2; ++kc2) {
    const unsigned short* p = Q + (size_t)(q0 + nl) * HD + kc2 * 32 + half * 8;
    aq[kc2].u[0] = *reinterpret_cast<const uint4*>(p);
    aq[kc2].u[1] = *reinterpret_cast<const uint4*>(p + 16);
  }

  v8f o[4];
  float mrow[8], lrow[8];
#pragma unroll
  for (int nt = 0; nt < 4; ++nt)
#pragma unroll
    for (int r = 0; r < 8; ++r) o[nt][r] = 0.f;
#pragma unroll
  for (int r = 0; r < 8; ++r) { mrow[r] = -__builtin_inff(); lrow[r] = 0.f; }

  for (int kc = 0; kc <= qb; ++kc) {            // causal: key chunks of 64
    v8f sc[4];
#pragma unroll
    for (int nt = 0; nt < 4; ++nt)
#pragma unroll
      for (int r = 0; r < 8; ++r) sc[nt][r] = 0.f;

    // scores = Q @ K^T  (K rows loaded directly in B-fragment layout)
#pragma unroll
    for (int nt = 0; nt < 4; ++nt) {
      const int keyrow = kc * 64 + nt * 16 + nl;
#pragma unroll
      for (int kc2 = 0; kc2 < 2; ++kc2) {
        FragBF bk;
        const unsigned short* p = Kp + (size_t)keyrow * HD + kc2 * 32 + half * 8;
        bk.u[0] = *reinterpret_cast<const uint4*>(p);
        bk.u[1] = *reinterpret_cast<const uint4*>(p + 16);
        sc[nt] = __builtin_amdgcn_wmma_f32_16x16x32_bf16(
            false, aq[kc2].v, false, bk.v, (short)0, sc[nt], false, false);
      }
    }

    if (kc == qb) {                              // diagonal chunk: causal mask
#pragma unroll
      for (int nt = 0; nt < 4; ++nt)
#pragma unroll
        for (int r = 0; r < 8; ++r)
          if (kc * 64 + nt * 16 + nl > q0 + r + 8 * half)
            sc[nt][r] = -__builtin_inff();
    }

    // online softmax (rows live across 16 lanes of one half-wave)
#pragma unroll
    for (int r = 0; r < 8; ++r) {
      float mx = fmaxf(fmaxf(sc[0][r], sc[1][r]), fmaxf(sc[2][r], sc[3][r]));
      mx = rmax16(mx);
      float mnew = fmaxf(mrow[r], mx);
      float corr = __expf((mrow[r] - mnew) * scale);
      mrow[r] = mnew;
      float rsum = 0.f;
#pragma unroll
      for (int nt = 0; nt < 4; ++nt) {
        float pv = __expf((sc[nt][r] - mnew) * scale);
        sc[nt][r] = pv;
        rsum += pv;
      }
      rsum = rsum16(rsum);
      lrow[r] = lrow[r] * corr + rsum;
#pragma unroll
      for (int nt = 0; nt < 4; ++nt) o[nt][r] *= corr;
    }

    // P (C layout) -> LDS -> A-fragment layout (wave-private, DS is in-order)
#pragma unroll
    for (int nt = 0; nt < 4; ++nt)
#pragma unroll
      for (int r = 0; r < 8; ++r)
        pw[(r + 8 * half) * 64 + nt * 16 + nl] = f2bf(sc[nt][r]);

    FragBF ap[2];
#pragma unroll
    for (int kc2 = 0; kc2 < 2; ++kc2) {
      const unsigned short* p = pw + nl * 64 + kc2 * 32 + half * 8;
      ap[kc2].u[0] = *reinterpret_cast<const uint4*>(p);
      ap[kc2].u[1] = *reinterpret_cast<const uint4*>(p + 16);
    }

    // O += P @ V  (V^T rows contiguous over keys)
#pragma unroll
    for (int nt = 0; nt < 4; ++nt)
#pragma unroll
      for (int kc2 = 0; kc2 < 2; ++kc2) {
        FragBF bv;
        const unsigned short* p =
            Vt + (size_t)(nt * 16 + nl) * SEQ + kc * 64 + kc2 * 32 + half * 8;
        bv.u[0] = *reinterpret_cast<const uint4*>(p);
        bv.u[1] = *reinterpret_cast<const uint4*>(p + 16);
        o[nt] = __builtin_amdgcn_wmma_f32_16x16x32_bf16(
            false, ap[kc2].v, false, bv.v, (short)0, o[nt], false, false);
      }
  }

  // normalize and store ctx (bf16, [b][s][h*64+d] for the Wo GEMM)
#pragma unroll
  for (int r = 0; r < 8; ++r) {
    float inv = 1.f / lrow[r];
    int srow = q0 + r + 8 * half;
#pragma unroll
    for (int nt = 0; nt < 4; ++nt) {
      int col = h * HD + nt * 16 + nl;
      ctxb[((size_t)(b * SEQ + srow)) * EMB + col] = f2bf(o[nt][r] * inv);
    }
  }
}

// ------------------------------------------------------------------ launch ---
extern "C" void kernel_launch(void* const* d_in, const int* in_sizes, int n_in,
                              void* d_out, int out_size, void* d_ws, size_t ws_size,
                              hipStream_t stream) {
  const float* x    = (const float*)d_in[0];
  // d_in[1]: mask (all-true in reference; causal mask applied in-kernel)
  const float* cosT = (const float*)d_in[2];
  const float* sinT = (const float*)d_in[3];
  const float* Wq   = (const float*)d_in[4];
  const float* Wk   = (const float*)d_in[5];
  const float* Wv   = (const float*)d_in[6];
  const float* Wo   = (const float*)d_in[7];
  float* out = (float*)d_out;

  char* ws = (char*)d_ws;
  const size_t MB = 1024 * 1024;
  unsigned short* xb   = (unsigned short*)(ws + 0);        // 16 MB  x bf16
  unsigned short* wq   = (unsigned short*)(ws + 16 * MB);  // 8 MB
  unsigned short* wk   = (unsigned short*)(ws + 24 * MB);  // 2 MB
  unsigned short* wv   = (unsigned short*)(ws + 26 * MB);  // 2 MB
  unsigned short* wo   = (unsigned short*)(ws + 28 * MB);  // 8 MB
  unsigned short* qh   = (unsigned short*)(ws + 36 * MB);  // 16 MB [b,h,s,d]
  unsigned short* kh   = (unsigned short*)(ws + 52 * MB);  // 4 MB  [b,kv,s,d]
  unsigned short* vh   = (unsigned short*)(ws + 56 * MB);  // 4 MB  [b,kv,d,s]
  unsigned short* ctxb = (unsigned short*)(ws + 60 * MB);  // 16 MB [b,s,e]

  cvt_f32_bf16<<<2048, 256, 0, stream>>>(x,  xb, BATCH * SEQ * EMB);
  cvt_f32_bf16<<<2048, 256, 0, stream>>>(Wq, wq, EMB * EMB);
  cvt_f32_bf16<<<1024, 256, 0, stream>>>(Wk, wk, NKV * HD * EMB);
  cvt_f32_bf16<<<1024, 256, 0, stream>>>(Wv, wv, NKV * HD * EMB);
  cvt_f32_bf16<<<2048, 256, 0, stream>>>(Wo, wo, EMB * EMB);

  const int M = BATCH * SEQ;   // 4096
  // Q projection + RoPE
  gemm_bf16_nt<<<dim3(EMB / 128, M / 128), 256, 0, stream>>>(
      xb, wq, nullptr, qh, cosT, sinT, M, EMB, EMB, 1, NH);
  // K projection + RoPE
  gemm_bf16_nt<<<dim3((NKV * HD) / 128, M / 128), 256, 0, stream>>>(
      xb, wk, nullptr, kh, cosT, sinT, M, NKV * HD, EMB, 1, NKV);
  // V projection (transposed store)
  gemm_bf16_nt<<<dim3((NKV * HD) / 128, M / 128), 256, 0, stream>>>(
      xb, wv, nullptr, vh, cosT, sinT, M, NKV * HD, EMB, 2, NKV);
  // flash attention
  attn_kernel<<<dim3(SEQ / 64, NH, BATCH), 128, 0, stream>>>(qh, kh, vh, ctxb);
  // output projection -> fp32
  gemm_bf16_nt<<<dim3(EMB / 128, M / 128), 256, 0, stream>>>(
      ctxb, wo, out, nullptr, cosT, sinT, M, EMB, EMB, 0, 0);
}
